// FusionRQVAE_47923245089217
// MI455X (gfx1250) — compile-verified, matrix operand order checked by reading, not run on hardware
//
#include <hip/hip_runtime.h>
#include <hip/hip_bf16.h>

// CDNA5 / gfx1250: wave32, WMMA 16x16x32 bf16 with f32 accumulate.
typedef __attribute__((ext_vector_type(16))) __bf16 v16bf;
typedef __attribute__((ext_vector_type(8)))  float  v8f;

#define BM 256
#define BN 64
#define BK 32

// Pack two f32 -> packed bf16 pair (compiler emits v_cvt_pk_bf16_f32).
__device__ __forceinline__ unsigned int pk2(float a, float b) {
  union { unsigned int u; __bf16 h[2]; } r;
  r.h[0] = (__bf16)a; r.h[1] = (__bf16)b;
  return r.u;
}

// Fragment-order slot for k within a 32-wide chunk (swap bits 3 and 4):
// half-0 lanes read slots [0,16) = k {0..7,16..23}; half-1 read [16,32) = k {8..15,24..31}.
__device__ __forceinline__ int fslot(int k) {
  return (k & 7) | ((k & 8) << 1) | ((k & 16) >> 1);
}

// ---------------------------------------------------------------------------
// GEMM: C[M x Ncols] = act(A[M x K] * W[K x Ncols] + bias [+ resid])
// 256x64 block tile, 8 waves, each wave a 32x64 patch = 2x4 WMMA tiles
// (8 WMMA per k-tile vs 12 LDS b128 fragment loads, 1 barrier per 8 WMMA).
// LDS tiles stored in WMMA-fragment order; double-buffered.
// Requires M % 256 == 0, K % 32 == 0; Ncols arbitrary (guarded).
// ---------------------------------------------------------------------------
__global__ __launch_bounds__(256) void gemm_bias_act_wmma(
    const float* __restrict__ A, const float* __restrict__ W,
    const float* __restrict__ bias, const float* __restrict__ resid,
    float* __restrict__ C, int M, int K, int Ncols, int relu)
{
  __shared__ __bf16 As[2][BM][40];   // rows in fragment order, pad 40 (80B stride) -> 40 KB
  __shared__ __bf16 Bs[2][BN][40];   // columns in fragment order               -> 10 KB

  const int tid  = threadIdx.x;
  const int lane = tid & 31;
  const int wave = tid >> 5;          // 8 row groups of 32
  const int half = lane >> 4;
  const int r    = lane & 15;
  const int rowBase = blockIdx.x * BM;
  const int colBase = blockIdx.y * BN;

  // staging assignments: A row = tid (full 32-k row), B col + 8-k group
  const int bcol  = tid & 63;
  const int bkg   = (tid >> 6) << 3;     // 0,8,16,24
  const int bslot = fslot(bkg);
  const bool bvalid = (colBase + bcol) < Ncols;

  auto stage = [&](int kt, int buf) {
    // A: one row per thread, 8x float4 -> 4x packed uint4 LDS stores
    const float* arow = A + (size_t)(rowBase + tid) * K + kt;
#pragma unroll
    for (int g = 0; g < 4; ++g) {
      const float4 v0 = *(const float4*)(arow + g * 8);
      const float4 v1 = *(const float4*)(arow + g * 8 + 4);
      uint4 p;
      p.x = pk2(v0.x, v0.y); p.y = pk2(v0.z, v0.w);
      p.z = pk2(v1.x, v1.y); p.w = pk2(v1.z, v1.w);
      *(uint4*)&As[buf][tid][fslot(g * 8)] = p;
    }
    // B: one column, 8 k values (wave-coalesced 4B loads), one 16B store
    float bv[8];
#pragma unroll
    for (int i = 0; i < 8; ++i)
      bv[i] = bvalid ? W[(size_t)(kt + bkg + i) * Ncols + (colBase + bcol)] : 0.f;
    uint4 p4;
    p4.x = pk2(bv[0], bv[1]); p4.y = pk2(bv[2], bv[3]);
    p4.z = pk2(bv[4], bv[5]); p4.w = pk2(bv[6], bv[7]);
    *(uint4*)&Bs[buf][bcol][bslot] = p4;
  };

  v8f acc[2][4];
#pragma unroll
  for (int im = 0; im < 2; ++im)
#pragma unroll
    for (int in = 0; in < 4; ++in)
      acc[im][in] = (v8f){0.f,0.f,0.f,0.f,0.f,0.f,0.f,0.f};

  const int ntiles = K / BK;
  stage(0, 0);
  __syncthreads();

  for (int t = 0; t < ntiles; ++t) {
    if (t + 1 < ntiles) stage((t + 1) * BK, (t + 1) & 1);  // prefetch next tile
    const int cur = t & 1;

    v16bf af[2], bfg[4];
#pragma unroll
    for (int im = 0; im < 2; ++im)
      af[im] = *(const v16bf*)&As[cur][wave * 32 + im * 16 + r][half * 16];
#pragma unroll
    for (int in = 0; in < 4; ++in)
      bfg[in] = *(const v16bf*)&Bs[cur][in * 16 + r][half * 16];

#pragma unroll
    for (int im = 0; im < 2; ++im)
#pragma unroll
      for (int in = 0; in < 4; ++in)
        acc[im][in] = __builtin_amdgcn_wmma_f32_16x16x32_bf16(
            false, af[im], false, bfg[in], (short)0, acc[im][in], false, false);
    __syncthreads();
  }

  // epilogue: C/D layout M = v + 8*half, N = r
#pragma unroll
  for (int im = 0; im < 2; ++im) {
#pragma unroll
    for (int in = 0; in < 4; ++in) {
      const int col = colBase + in * 16 + r;
      if (col >= Ncols) continue;
      const float bv = bias ? bias[col] : 0.f;
#pragma unroll
      for (int v = 0; v < 8; ++v) {
        const int row = rowBase + wave * 32 + im * 16 + v + (half << 3);
        float x = acc[im][in][v] + bv;
        if (resid) x += resid[(size_t)row * Ncols + col];
        if (relu)  x = fmaxf(x, 0.f);
        C[(size_t)row * Ncols + col] = x;
      }
    }
  }
}

// ---------------------------------------------------------------------------
// Fused residual VQ (L=4, K=256, E=64). One wave per 16 items.
// Distances r.cb^T via WMMA from fragment-ordered LDS; running per-register
// argmin across 16 code tiles; one shfl-butterfly argmin at level end.
// 2 block barriers per level.
// ---------------------------------------------------------------------------
#define RQW 8

__global__ __launch_bounds__(256) void rq_wmma_kernel(
    const float* __restrict__ zf,     // [N][64]
    const float* __restrict__ cb,     // [4][256][64]
    float* __restrict__ zq,           // [N][64]
    int*   __restrict__ idx_out,      // [N][4]
    float* __restrict__ loss_accum,   // 1 float
    int Nitems)
{
  __shared__ __bf16 cb_frag[256][72];        // fragment-ordered, pad 72 -> 36 KB
  __shared__ float  c2[256];                 // 1 KB
  __shared__ __bf16 res_frag[RQW][16][72];   // 18 KB
  __shared__ int    idxm[RQW][16];           // per-wave argmin exchange

  const int tid  = threadIdx.x;
  const int lane = tid & 31;
  const int wave = tid >> 5;
  const int half = lane >> 4;
  const int r    = lane & 15;
  const int tileBase = (blockIdx.x * RQW + wave) * 16;
  const int item = tileBase + r;             // lane pair (r, r+16) shares item r

  // lane holds its item's elements e = half*32 + j, j = 0..31
  float res[32], xq[32];
  {
    const float* zrow = zf + (size_t)item * 64 + half * 32;
#pragma unroll
    for (int j = 0; j < 32; j += 4) {
      const float4 v = *(const float4*)(zrow + j);
      res[j] = v.x; res[j+1] = v.y; res[j+2] = v.z; res[j+3] = v.w;
      xq[j] = 0.f; xq[j+1] = 0.f; xq[j+2] = 0.f; xq[j+3] = 0.f;
    }
  }
  float loss_sum = 0.f;

  for (int l = 0; l < 4; ++l) {
    const float* cbl = cb + (size_t)l * 256 * 64;
    // stage codebook row `tid` in fragment order + ||c||^2
    {
      float s = 0.f;
      const float* crow = cbl + (size_t)tid * 64;
#pragma unroll
      for (int c = 0; c < 2; ++c)
#pragma unroll
        for (int g = 0; g < 4; ++g) {
          const int e0 = c * 32 + g * 8;
          const float4 v0 = *(const float4*)(crow + e0);
          const float4 v1 = *(const float4*)(crow + e0 + 4);
          uint4 p;
          p.x = pk2(v0.x, v0.y); p.y = pk2(v0.z, v0.w);
          p.z = pk2(v1.x, v1.y); p.w = pk2(v1.z, v1.w);
          *(uint4*)&cb_frag[tid][c * 32 + fslot(g * 8)] = p;
          s += v0.x*v0.x + v0.y*v0.y + v0.z*v0.z + v0.w*v0.w
             + v1.x*v1.x + v1.y*v1.y + v1.z*v1.z + v1.w*v1.w;
        }
      c2[tid] = s;
    }
    // stage residual in fragment order (chunk = half)
#pragma unroll
    for (int g = 0; g < 4; ++g) {
      const int j0 = g * 8;
      uint4 p;
      p.x = pk2(res[j0+0], res[j0+1]); p.y = pk2(res[j0+2], res[j0+3]);
      p.z = pk2(res[j0+4], res[j0+5]); p.w = pk2(res[j0+6], res[j0+7]);
      *(uint4*)&res_frag[wave][r][half * 32 + fslot(j0)] = p;
    }
    __syncthreads();   // (1) staging visible

    v16bf af0 = *(const v16bf*)&res_frag[wave][r][half * 16];
    v16bf af1 = *(const v16bf*)&res_frag[wave][r][32 + half * 16];

    float minv[8]; int mini[8];
#pragma unroll
    for (int v = 0; v < 8; ++v) { minv[v] = 3.4e38f; mini[v] = 0; }

    for (int ct = 0; ct < 16; ++ct) {
      const int code = ct * 16 + r;      // lane owns one code column
      const v16bf b0 = *(const v16bf*)&cb_frag[code][half * 16];
      const v16bf b1 = *(const v16bf*)&cb_frag[code][32 + half * 16];
      v8f acc = (v8f){0.f,0.f,0.f,0.f,0.f,0.f,0.f,0.f};
      acc = __builtin_amdgcn_wmma_f32_16x16x32_bf16(false, af0, false, b0, (short)0, acc, false, false);
      acc = __builtin_amdgcn_wmma_f32_16x16x32_bf16(false, af1, false, b1, (short)0, acc, false, false);
      const float c2c = c2[code];
#pragma unroll
      for (int v = 0; v < 8; ++v) {      // item v + 8*half vs this lane's code
        const float d = c2c - 2.f * acc[v];
        if (d < minv[v] || (d == minv[v] && code < mini[v])) { minv[v] = d; mini[v] = code; }
      }
    }
    // butterfly argmin across the 16 lanes of each half (codes)
#pragma unroll
    for (int v = 0; v < 8; ++v) {
#pragma unroll
      for (int m = 8; m; m >>= 1) {
        const float od = __shfl_xor(minv[v], m, 32);
        const int   oi = __shfl_xor(mini[v], m, 32);
        if (od < minv[v] || (od == minv[v] && oi < mini[v])) { minv[v] = od; mini[v] = oi; }
      }
    }
    // exchange: lane with (lane&15)==v in each half publishes item v+8*half
#pragma unroll
    for (int v = 0; v < 8; ++v)
      if ((lane & 15) == v) idxm[wave][v + 8 * half] = mini[v];
    asm volatile("s_wait_dscnt 0" ::: "memory");   // same-wave LDS is in-order
    const int myidx = idxm[wave][lane & 15];

    // straight-through update with f32 codebook gather
    const float* qrow = cbl + (size_t)myidx * 64 + half * 32;
    float lsum = 0.f;
#pragma unroll
    for (int j = 0; j < 32; j += 4) {
      const float4 q = *(const float4*)(qrow + j);
      const float d0 = q.x - res[j],   d1 = q.y - res[j+1];
      const float d2 = q.z - res[j+2], d3 = q.w - res[j+3];
      lsum += d0*d0 + d1*d1 + d2*d2 + d3*d3;
      xq[j]   += q.x; xq[j+1] += q.y; xq[j+2] += q.z; xq[j+3] += q.w;
      res[j]  -= q.x; res[j+1] -= q.y; res[j+2] -= q.z; res[j+3] -= q.w;
    }
    loss_sum += lsum;
    if (half == 0 && item < Nitems) idx_out[(size_t)item * 4 + l] = myidx;
    __syncthreads();   // (2) protect cb_frag/c2 before next level restage
  }

  for (int off = 16; off; off >>= 1) loss_sum += __shfl_xor(loss_sum, off, 32);
  if (lane == 0) atomicAdd(loss_accum, loss_sum);

  float* zqrow = zq + (size_t)item * 64 + half * 32;
#pragma unroll
  for (int j = 0; j < 32; j += 4) {
    const float4 v = {xq[j], xq[j+1], xq[j+2], xq[j+3]};
    *(float4*)(zqrow + j) = v;
  }
}

__global__ void finalize_loss_kernel(const float* acc, float* out_t, float* out_v, float scale) {
  if (threadIdx.x == 0) {
    *out_t = acc[0] * scale;
    *out_v = acc[1] * scale;
  }
}

// ---------------------------------------------------------------------------
// Orchestration
// ---------------------------------------------------------------------------
extern "C" void kernel_launch(void* const* d_in, const int* in_sizes, int n_in,
                              void* d_out, int out_size, void* d_ws, size_t ws_size,
                              hipStream_t stream) {
  const int N = in_sizes[0] / 768;   // 65536
  auto P = [&](int i) { return (const float*)d_in[i]; };

  // d_in: 0:x_text 1:x_vis | 2..9 text_enc | 10..17 vis_enc
  //       18..25 text_dec | 26..33 vis_dec | 34 text_cb 35 vis_cb
  //       36 text_A 37 text_B 38 vis_A 39 vis_B
  const float* x_text = P(0);
  const float* x_vis  = P(1);

  float* ws = (float*)d_ws;
  size_t off = 0;
  float* buf0    = ws + off; off += (size_t)N * 1024;
  float* buf1    = ws + off; off += (size_t)N * 512;
  float* z_text  = ws + off; off += (size_t)N * 64;
  float* z_vis   = ws + off; off += (size_t)N * 64;
  float* zf_text = ws + off; off += (size_t)N * 64;
  float* zf_vis  = ws + off; off += (size_t)N * 64;
  float* zq_text = ws + off; off += (size_t)N * 64;
  float* zq_vis  = ws + off; off += (size_t)N * 64;
  float* tmp     = ws + off; off += (size_t)N * 32;
  float* lacc    = ws + off; off += 2;

  float* out      = (float*)d_out;
  float* out_text = out;
  float* out_vis  = out_text + (size_t)N * 768;
  float* loss_t   = out_vis + (size_t)N * 512;
  float* loss_v   = loss_t + 1;
  int*   idx_t    = (int*)(loss_v + 1);
  int*   idx_v    = idx_t + (size_t)N * 4;

  auto gemm = [&](const float* A, const float* W, const float* b,
                  const float* resid, float* C, int K, int Nc, int relu) {
    dim3 grid((N + BM - 1) / BM, (Nc + BN - 1) / BN);
    gemm_bias_act_wmma<<<grid, 256, 0, stream>>>(A, W, b, resid, C, N, K, Nc, relu);
  };

  // encoders
  gemm(x_text, P(2), P(3), nullptr, buf0,   768, 1024, 1);
  gemm(buf0,   P(4), P(5), nullptr, buf1,  1024,  512, 1);
  gemm(buf1,   P(6), P(7), nullptr, buf0,   512,  256, 1);
  gemm(buf0,   P(8), P(9), nullptr, z_text,  256,   64, 0);

  gemm(x_vis, P(10), P(11), nullptr, buf0,   512, 1024, 1);
  gemm(buf0,  P(12), P(13), nullptr, buf1,  1024,  512, 1);
  gemm(buf1,  P(14), P(15), nullptr, buf0,   512,  256, 1);
  gemm(buf0,  P(16), P(17), nullptr, z_vis,  256,   64, 0);

  // LoRA cross-modal fusion (ALPHA=1 folded into residual add)
  gemm(z_vis,  P(39), nullptr, nullptr, tmp,     64, 32, 0);  // z_vis @ vis_B
  gemm(tmp,    P(36), nullptr, z_text,  zf_text, 32, 64, 0);  // + @ text_A
  gemm(z_text, P(37), nullptr, nullptr, tmp,     64, 32, 0);  // z_text @ text_B
  gemm(tmp,    P(38), nullptr, z_vis,   zf_vis,  32, 64, 0);  // + @ vis_A

  // residual VQ
  hipMemsetAsync(lacc, 0, 2 * sizeof(float), stream);
  const int rq_blocks = N / (16 * RQW);
  rq_wmma_kernel<<<rq_blocks, 256, 0, stream>>>(zf_text, P(34), zq_text, idx_t, lacc + 0, N);
  rq_wmma_kernel<<<rq_blocks, 256, 0, stream>>>(zf_vis,  P(35), zq_vis,  idx_v, lacc + 1, N);

  // decoders
  gemm(zq_text, P(18), P(19), nullptr, buf0,      64,  256, 1);
  gemm(buf0,    P(20), P(21), nullptr, buf1,     256,  512, 1);
  gemm(buf1,    P(22), P(23), nullptr, buf0,     512, 1024, 1);
  gemm(buf0,    P(24), P(25), nullptr, out_text, 1024, 768, 0);

  gemm(zq_vis, P(26), P(27), nullptr, buf0,      64,  256, 1);
  gemm(buf0,   P(28), P(29), nullptr, buf1,     256,  512, 1);
  gemm(buf1,   P(30), P(31), nullptr, buf0,     512, 1024, 1);
  gemm(buf0,   P(32), P(33), nullptr, out_vis, 1024,  512, 0);

  const float scale = (1.0f + 0.25f) / (4.0f * (float)N * 64.0f);
  finalize_loss_kernel<<<1, 1, 0, stream>>>(lacc, loss_t, loss_v, scale);
}
